// MultiClassIoU_90374701843006
// MI455X (gfx1250) — 8-bit matrix operand order repaired and checked by reading, compiled, not run on hardware
//
#include <hip/hip_runtime.h>
#include <math.h>


#define NUM_CLASSES 21
#define HW          (512 * 512)
#define BATCH       16
#define TOTAL_PIX   (BATCH * HW)
#define CONF_DIM    32

#define WAVES_PER_BLOCK 8
#define THREADS         (WAVES_PER_BLOCK * 32)
#define BLOCKS          2048
#define TOTAL_WAVES     (BLOCKS * WAVES_PER_BLOCK)
#define TOTAL_CHUNKS    (TOTAL_PIX / 64)
#define CHUNKS_PER_WAVE (TOTAL_CHUNKS / TOTAL_WAVES)

typedef __attribute__((ext_vector_type(8))) int   v8i;
typedef __attribute__((ext_vector_type(2))) float f32x2;

__global__ void MultiClassIoU_zero_kernel(int* __restrict__ conf) {
    int i = threadIdx.x;
    if (i < CONF_DIM * CONF_DIM) conf[i] = 0;
}

__device__ __forceinline__ unsigned onehot4(unsigned x, unsigned rrrr) {
    unsigned y = x ^ rrrr;
    return ((0x40404040u - y) & 0x40404040u) >> 6;
}

__device__ __forceinline__ void wave_lds_fence() {
    __builtin_amdgcn_fence(__ATOMIC_ACQ_REL, "wavefront");
}

__global__ __launch_bounds__(THREADS)
void MultiClassIoU_conf_kernel(const float* __restrict__ pred,
                               const int*   __restrict__ tgt,
                               int*         __restrict__ conf) {
    __shared__ unsigned char s_pred[WAVES_PER_BLOCK][64];
    __shared__ unsigned char s_tgt [WAVES_PER_BLOCK][64];

    const int lane  = threadIdx.x & 31;
    const int wave  = threadIdx.x >> 5;
    const int gwave = blockIdx.x * WAVES_PER_BLOCK + wave;

    v8i c00 = {}; v8i c01 = {}; v8i c10 = {}; v8i c11 = {};

    const int half = lane >> 4;
    const int lid  = lane & 15;
    const unsigned r0 = 0x01010101u * (unsigned)lid;
    const unsigned r1 = 0x01010101u * (unsigned)(lid + 16);

    #pragma unroll
    for (int i = 0; i < CHUNKS_PER_WAVE; ++i) {
        const int       chunk   = gwave * CHUNKS_PER_WAVE + i;
        const long long pixBase = (long long)chunk * 64;
        const int       b       = (int)(pixBase / HW);
        const int       pixInB  = (int)(pixBase % HW) + 2 * lane;

        const float* base = pred + ((long long)b * NUM_CLASSES) * HW + pixInB;
        float best0 = -INFINITY, best1 = -INFINITY;
        int   cls0 = 0, cls1 = 0;
        #pragma unroll
        for (int c = 0; c < NUM_CLASSES; ++c) {
            f32x2 v = __builtin_nontemporal_load(
                          (const f32x2*)(base + (long long)c * HW));
            bool g0 = v.x > best0;
            bool g1 = v.y > best1;
            cls0  = g0 ? c   : cls0;  best0 = g0 ? v.x : best0;
            cls1  = g1 ? c   : cls1;  best1 = g1 ? v.y : best1;
        }
        const long long ti = pixBase + 2 * lane;
        const int t0 = __builtin_nontemporal_load(tgt + ti);
        const int t1 = __builtin_nontemporal_load(tgt + ti + 1);

        *(unsigned short*)&s_pred[wave][2 * lane] =
            (unsigned short)(cls0 | (cls1 << 8));
        *(unsigned short*)&s_tgt[wave][2 * lane] =
            (unsigned short)(t0 | (t1 << 8));
        wave_lds_fence();

        const unsigned* pp = (const unsigned*)s_pred[wave];
        const unsigned* tp = (const unsigned*)s_tgt[wave];

        v8i a0, a1;
        #pragma unroll
        for (int v = 0; v < 8; ++v) {
            int k = ((v >> 2) & 1) * 32 + ((v >> 1) & 1) * 16 +
                    (v & 1) * 4 + half * 8;
            unsigned x = pp[k >> 2];
            a0[v] = (int)onehot4(x, r0);
            a1[v] = (int)onehot4(x, r1);
        }
        v8i b0, b1;
        #pragma unroll
        for (int v = 0; v < 8; ++v) {
            int k = ((v >> 2) & 1) * 32 + ((v >> 1) & 1) * 16 + (v & 1) * 4 + half * 8;
            unsigned x = tp[k >> 2];
            b0[v] = (int)onehot4(x, r0);
            b1[v] = (int)onehot4(x, r1);
        }
        wave_lds_fence();

        c00 = __builtin_amdgcn_wmma_i32_16x16x64_iu8(false, a0, false, b0, c00, false, false);
        c01 = __builtin_amdgcn_wmma_i32_16x16x64_iu8(false, a0, false, b1, c01, false, false);
        c10 = __builtin_amdgcn_wmma_i32_16x16x64_iu8(false, a1, false, b0, c10, false, false);
        c11 = __builtin_amdgcn_wmma_i32_16x16x64_iu8(false, a1, false, b1, c11, false, false);
    }

    const int Mbase = (lane >> 4) * 8;
    const int N     = lane & 15;
    #pragma unroll
    for (int p = 0; p < 8; ++p) {
        const int m = Mbase + p;
        atomicAdd(&conf[m        * CONF_DIM + N],        c00[p]);
        atomicAdd(&conf[m        * CONF_DIM + (N + 16)], c01[p]);
        atomicAdd(&conf[(m + 16) * CONF_DIM + N],        c10[p]);
        atomicAdd(&conf[(m + 16) * CONF_DIM + (N + 16)], c11[p]);
    }
}

__global__ void MultiClassIoU_finalize_kernel(const int* __restrict__ conf,
                                              float* __restrict__ out) {
    const int c = threadIdx.x;
    float iou = 0.0f;
    if (c < NUM_CLASSES) {
        float inter = (float)conf[c * CONF_DIM + c];
        float pcnt = 0.0f, tcnt = 0.0f;
        #pragma unroll
        for (int j = 0; j < CONF_DIM; ++j) {
            pcnt += (float)conf[c * CONF_DIM + j];
            tcnt += (float)conf[j * CONF_DIM + c];
        }
        float uni = pcnt + tcnt - inter;
        iou = (uni > 0.0f) ? inter / (uni + 1e-6f) : 0.0f;
    }
    #pragma unroll
    for (int off = 16; off > 0; off >>= 1)
        iou += __shfl_down(iou, off, 32);
    if (c == 0) out[0] = iou / (float)NUM_CLASSES;
}

extern "C" void kernel_launch(void* const* d_in, const int* in_sizes, int n_in,
                              void* d_out, int out_size, void* d_ws, size_t ws_size,
                              hipStream_t stream) {
    const float* pred = (const float*)d_in[0];
    const int*   tgt  = (const int*)  d_in[1];
    float*       out  = (float*)d_out;
    int*         conf = (int*)d_ws;

    MultiClassIoU_zero_kernel<<<1, 1024, 0, stream>>>(conf);
    MultiClassIoU_conf_kernel<<<BLOCKS, THREADS, 0, stream>>>(pred, tgt, conf);
    MultiClassIoU_finalize_kernel<<<1, 32, 0, stream>>>(conf, out);
}
